// NoisyPositionEmbedder_47502338294327
// MI455X (gfx1250) — matrix operand order repaired and checked by reading, not compiled, not run on hardware
//
#include <hip/hip_runtime.h>

// CDNA5 / gfx1250. wave32. WMMA f32_16x16x32_f16 (codegen-confirmed builtin).
typedef __attribute__((ext_vector_type(16))) _Float16 v16h;
typedef __attribute__((ext_vector_type(8)))  float    v8f;

#define EPSF 1e-5f
#define A_DIM 1024
#define T_DIM 256
#define CS 384
#define CZ 128
#define CATOM 128
#define CPAIR 16

__device__ __forceinline__ float wave_reduce_add(float v) {
  // wave32 butterfly reduction
  #pragma unroll
  for (int off = 16; off > 0; off >>= 1) v += __shfl_xor(v, off, 32);
  return v;
}

// ---------------- prep kernels ----------------

// Recover integer token index from one-hot rows of a2t.
__global__ void tok_kernel(const float* __restrict__ a2t, int* __restrict__ tok) {
  int a = blockIdx.x * blockDim.x + threadIdx.x;
  if (a >= A_DIM) return;
  const float* row = a2t + (size_t)a * T_DIM;
  int t = 0;
  for (int j = 0; j < T_DIM; ++j) if (row[j] > 0.5f) t = j;
  tok[a] = t;
}

// Bake B-matrix WMMA fragments for W' = diag(g) @ W  (K x N, f16), in
// per-lane fragment order: Bf[(nt*chunks + c)*32 + lane][e], e=0..15.
// B layout (16-bit 32x16): lane L: n = L%16, h = L/16; element e <-> K = 32c + 8h + (e&7) + 16*(e>>3).
__global__ void prep_bfrag(const float* __restrict__ g, const float* __restrict__ W,
                           _Float16* __restrict__ Bf, int K, int N) {
  const int chunks = K >> 5;
  const int tid = blockIdx.x * blockDim.x + threadIdx.x;
  const int total = (N >> 4) * chunks * 32;
  if (tid >= total) return;
  const int lane  = tid & 31;
  const int c     = (tid >> 5) % chunks;
  const int nt    = tid / (32 * chunks);
  const int n     = lane & 15;
  const int kbase = (lane >> 4) << 4;
  _Float16* dst = Bf + (size_t)tid * 16;
  #pragma unroll 4
  for (int e = 0; e < 16; ++e) {
    const int k = (c << 5) + (kbase >> 1) + (e & 7) + ((e >> 3) << 4);
    dst[e] = (_Float16)(g[k] * W[(size_t)k * N + nt * 16 + n]);
  }
}

// bias_out[n] = sum_k b[k] * W[k,n]
__global__ void prep_bias(const float* __restrict__ b, const float* __restrict__ W,
                          float* __restrict__ out, int K, int N) {
  int n = blockIdx.x * blockDim.x + threadIdx.x;
  if (n >= N) return;
  float s = 0.f;
  for (int k = 0; k < K; ++k) s += b[k] * W[(size_t)k * N + n];
  out[n] = s;
}

// ---------------- ql: trivial rank-3 update ----------------
__global__ void ql_kernel(const float* __restrict__ ql, const float* __restrict__ rl,
                          const float* __restrict__ Wr, float* __restrict__ out) {
  int i = blockIdx.x * blockDim.x + threadIdx.x;
  if (i >= A_DIM * CATOM) return;
  int a = i >> 7, c = i & 127;
  float s = ql[i];
  #pragma unroll
  for (int k = 0; k < 3; ++k) s += rl[a * 3 + k] * Wr[k * CATOM + c];
  out[i] = s;
}

// ---------------- cl: gather + LN + [1024x384]@[384x128] via WMMA ----------------
// 64 blocks of 8 waves. Block handles one 16-row M-tile; wave w handles N-tile w.
__global__ __launch_bounds__(256) void cl_kernel(
    const float* __restrict__ cl, const float* __restrict__ s_trunk,
    const int* __restrict__ tok, const _Float16* __restrict__ Bfs,
    const float* __restrict__ bias_s, float* __restrict__ out) {
  __shared__ _Float16 lds[16 * CS];  // normalized f16 tile, 12 KB
  const int lane  = threadIdx.x & 31;
  const int wave  = threadIdx.x >> 5;
  const int mtile = blockIdx.x;

  // Stage: each wave normalizes 2 of the 16 rows (LN over 384).
  #pragma unroll
  for (int r = 0; r < 2; ++r) {
    const int m = wave * 2 + r;
    const int a = mtile * 16 + m;
    const float* src = s_trunk + (size_t)tok[a] * CS;
    float4 v[3];
    #pragma unroll
    for (int j = 0; j < 3; ++j) v[j] = *(const float4*)(src + 4 * lane + 128 * j);
    float s = 0.f, ss = 0.f;
    #pragma unroll
    for (int j = 0; j < 3; ++j) {
      s  += v[j].x + v[j].y + v[j].z + v[j].w;
      ss += v[j].x * v[j].x + v[j].y * v[j].y + v[j].z * v[j].z + v[j].w * v[j].w;
    }
    s = wave_reduce_add(s); ss = wave_reduce_add(ss);
    const float mean = s * (1.f / CS);
    const float inv  = rsqrtf(ss * (1.f / CS) - mean * mean + EPSF);
    _Float16* dst = lds + m * CS;
    #pragma unroll
    for (int j = 0; j < 3; ++j) {
      const int k = 4 * lane + 128 * j;
      dst[k + 0] = (_Float16)((v[j].x - mean) * inv);
      dst[k + 1] = (_Float16)((v[j].y - mean) * inv);
      dst[k + 2] = (_Float16)((v[j].z - mean) * inv);
      dst[k + 3] = (_Float16)((v[j].w - mean) * inv);
    }
  }
  __syncthreads();

  // Compute: wave == N-tile. K = 384 -> 12 chained WMMAs.
  const int half_ = lane >> 4;
  const int mrow  = lane & 15;
  const int base  = half_ * 8;
  v8f acc = {};
  for (int c = 0; c < 12; ++c) {
    const _Float16* row = lds + mrow * CS + c * 32;
    v16h af;
    #pragma unroll
    for (int e = 0; e < 8; ++e) af[e] = row[base + e];
    #pragma unroll
    for (int e = 0; e < 8; ++e) af[8 + e] = row[16 + base + e];
    v16h bf = *(const v16h*)(Bfs + ((size_t)(wave * 12 + c) * 32 + lane) * 16);
    acc = __builtin_amdgcn_wmma_f32_16x16x32_f16(false, af, false, bf,
                                                 (short)0, acc, false, false);
  }
  const int n  = lane & 15;
  const float bv = bias_s[wave * 16 + n];
  #pragma unroll
  for (int v = 0; v < 8; ++v) {
    const int a = mtile * 16 + v + 8 * half_;
    const size_t idx = (size_t)a * CATOM + wave * 16 + n;
    out[idx] = cl[idx] + acc[v] + bv;
  }
}

// ---------------- plm: fused pair-gather + LN + [*x128]@[128x16] via WMMA ----------------
// 1M pairs -> 65536 tiles of 16 pairs. 8192 blocks x 8 waves; one tile per wave.
// zij (33.5 MB) stays L2-resident; HBM traffic ~= plm in+out (134 MB) -> BW-bound.
__global__ __launch_bounds__(256) void plm_kernel(
    const float* __restrict__ plm, const float* __restrict__ zij,
    const int* __restrict__ tok, const _Float16* __restrict__ Bfz,
    const float* __restrict__ biasz, float* __restrict__ out) {
  __shared__ _Float16 lds[8][16 * CZ];  // per-wave normalized tile, 4 KB each
  const int lane = threadIdx.x & 31;
  const int wave = threadIdx.x >> 5;
  const int tile = blockIdx.x * 8 + wave;     // 0..65535
  const int a    = tile >> 6;                 // 64 b-tiles per a
  const int b0   = (tile & 63) << 4;
  const int ta   = tok[a];
  _Float16* my = lds[wave];

  // Stage: gather zij[ta, tok[b0+m], :128], LN over 128, store f16.
  for (int m = 0; m < 16; ++m) {
    const int tb = tok[b0 + m];
    const float* src = zij + ((size_t)ta * T_DIM + tb) * CZ;
    const float4 v = *(const float4*)(src + 4 * lane);
    float s  = v.x + v.y + v.z + v.w;
    float ss = v.x * v.x + v.y * v.y + v.z * v.z + v.w * v.w;
    s = wave_reduce_add(s); ss = wave_reduce_add(ss);
    const float mean = s * (1.f / CZ);
    const float inv  = rsqrtf(ss * (1.f / CZ) - mean * mean + EPSF);
    _Float16* dst = my + m * CZ + 4 * lane;
    dst[0] = (_Float16)((v.x - mean) * inv);
    dst[1] = (_Float16)((v.y - mean) * inv);
    dst[2] = (_Float16)((v.z - mean) * inv);
    dst[3] = (_Float16)((v.w - mean) * inv);
  }
  // Wave-private LDS tile: enforce write->read ordering explicitly.
  asm volatile("s_wait_dscnt 0" ::: "memory");

  // K = 128 -> 4 chained WMMAs.
  const int half_ = lane >> 4;
  const int mrow  = lane & 15;
  const int base  = half_ * 8;
  v8f acc = {};
  #pragma unroll
  for (int c = 0; c < 4; ++c) {
    const _Float16* row = my + mrow * CZ + c * 32;
    v16h af;
    #pragma unroll
    for (int e = 0; e < 8; ++e) af[e] = row[base + e];
    #pragma unroll
    for (int e = 0; e < 8; ++e) af[8 + e] = row[16 + base + e];
    v16h bf = *(const v16h*)(Bfz + ((size_t)c * 32 + lane) * 16);
    acc = __builtin_amdgcn_wmma_f32_16x16x32_f16(false, af, false, bf,
                                                 (short)0, acc, false, false);
  }
  const int n = lane & 15;
  const float bv = biasz[n];
  #pragma unroll
  for (int v = 0; v < 8; ++v) {
    const size_t p   = (size_t)a * A_DIM + b0 + v + 8 * half_;
    const size_t idx = p * CPAIR + n;
    out[idx] = plm[idx] + acc[v] + bv;
  }
}

// ---------------- launch ----------------
extern "C" void kernel_launch(void* const* d_in, const int* in_sizes, int n_in,
                              void* d_out, int out_size, void* d_ws, size_t ws_size,
                              hipStream_t stream) {
  const float* a2t     = (const float*)d_in[0];
  const float* cl      = (const float*)d_in[1];
  const float* plm     = (const float*)d_in[2];
  const float* ql      = (const float*)d_in[3];
  const float* s_trunk = (const float*)d_in[4];
  const float* zij     = (const float*)d_in[5];
  const float* rl      = (const float*)d_in[6];
  const float* ln_s_g  = (const float*)d_in[7];
  const float* ln_s_b  = (const float*)d_in[8];
  const float* W_s     = (const float*)d_in[9];
  const float* ln_z_g  = (const float*)d_in[10];
  const float* ln_z_b  = (const float*)d_in[11];
  const float* W_z     = (const float*)d_in[12];
  const float* W_r     = (const float*)d_in[13];

  float* out     = (float*)d_out;
  float* out_cl  = out;                                   // 1024*128
  float* out_plm = out + A_DIM * CATOM;                   // 1024*1024*16
  float* out_ql  = out + A_DIM * CATOM + (size_t)A_DIM * A_DIM * CPAIR;

  char* ws = (char*)d_ws;
  int*      tok    = (int*)ws;                      // 4 KB
  float*    biasz  = (float*)(ws + 4096);           // 64 B
  float*    bias_s = (float*)(ws + 4608);           // 512 B
  _Float16* Bfz    = (_Float16*)(ws + 8192);        // 4*32*16 f16 = 4 KB
  _Float16* Bfs    = (_Float16*)(ws + 16384);       // 8*12*32*16 f16 = 96 KB

  tok_kernel<<<4, 256, 0, stream>>>(a2t, tok);
  prep_bfrag<<<1, 128, 0, stream>>>(ln_z_g, W_z, Bfz, CZ, CPAIR);
  prep_bfrag<<<12, 256, 0, stream>>>(ln_s_g, W_s, Bfs, CS, CATOM);
  prep_bias<<<1, CPAIR, 0, stream>>>(ln_z_b, W_z, biasz, CZ, CPAIR);
  prep_bias<<<1, CATOM, 0, stream>>>(ln_s_b, W_s, bias_s, CS, CATOM);

  ql_kernel<<<(A_DIM * CATOM) / 256, 256, 0, stream>>>(ql, rl, W_r, out_ql);
  cl_kernel<<<64, 256, 0, stream>>>(cl, s_trunk, tok, Bfs, bias_s, out_cl);
  plm_kernel<<<8192, 256, 0, stream>>>(plm, zij, tok, Bfz, biasz, out_plm);
}
